// MultiAttention_38431367364933
// MI455X (gfx1250) — compile-verified, matrix operand order checked by reading, not run on hardware
//
#include <hip/hip_runtime.h>

// ---------------- problem constants ----------------
#define BROWS 131072
#define NHEAD 10
#define DDIM  105

// ---------------- WMMA types ----------------
typedef __attribute__((ext_vector_type(16))) __bf16 v16bf;
typedef __attribute__((ext_vector_type(8)))  float  v8f;

union Frag {
  uint4 u[2];   // 32 bytes
  v16bf bf;     // 16 x bf16
};

__device__ __forceinline__ unsigned short f2bf(float f) {
  unsigned int u = __float_as_uint(f);
  unsigned int r = u + 0x7FFFu + ((u >> 16) & 1u);   // round-to-nearest-even
  return (unsigned short)(r >> 16);
}
__device__ __forceinline__ float bf2f(unsigned short b) {
  return __uint_as_float(((unsigned int)b) << 16);
}

// ---------------- packed-weight fragment geometry ----------------
// Every 16x16-output WMMA B fragment (32 K x 16 N, bf16) is stored as
// 32 lanes x 16 bf16 = 1024 bytes, in the exact per-lane order the
// v_wmma_f32_16x16x32_bf16 B operand wants:
//   lane l (l15=l&15, half=l>>4), element e:
//     N = ntile*16 + l15
//     K = ktile*32 + (e<8 ? e : e+8) + (half ? 8 : 0)
// Regions (frag index):
//   A  (z = ctx@A_w.T):  66 ntiles x 4 ktiles  -> frags [0,264)
//   HM:                   5 ntiles x 25 ktiles -> frags [264,389)
//   ME:                   2 ntiles x 7 ktiles  -> frags [389,403)
//   TF:                   1 ntile  x 2 ktiles  -> frags [403,405)
#define FRAGS_A_BASE  0
#define FRAGS_HM_BASE 264
#define FRAGS_ME_BASE 389
#define FRAGS_TF_BASE 403
#define FRAGS_TOTAL   405

__global__ void __launch_bounds__(256)
pack_weights_kernel(const float* __restrict__ A_w,
                    const float* __restrict__ HM_w,
                    const float* __restrict__ ME_w,
                    const float* __restrict__ TF_w,
                    unsigned short* __restrict__ wsb)
{
  int gid = blockIdx.x * 256 + threadIdx.x;
  if (gid >= FRAGS_TOTAL * 32) return;
  int frag = gid >> 5;
  int lane = gid & 31;
  int l15  = lane & 15;
  int half = lane >> 4;

  const float* W; int N, K, nt, kt;
  if (frag < FRAGS_HM_BASE)      { W = A_w;  N = 1050; K = 105; int f = frag;                 nt = f >> 2; kt = f & 3; }
  else if (frag < FRAGS_ME_BASE) { W = HM_w; N = 80;   K = 800; int f = frag - FRAGS_HM_BASE; nt = f / 25; kt = f % 25; }
  else if (frag < FRAGS_TF_BASE) { W = ME_w; N = 20;   K = 200; int f = frag - FRAGS_ME_BASE; nt = f / 7;  kt = f % 7; }
  else                           { W = TF_w; N = 5;    K = 50;  nt = 0; kt = frag - FRAGS_TF_BASE; }

  int n = nt * 16 + l15;
  unsigned short* outp = wsb + (size_t)frag * 512 + (size_t)lane * 16;
#pragma unroll
  for (int e = 0; e < 16; ++e) {
    int kp = (e < 8 ? e : e + 8) + (half ? 8 : 0);
    int k  = kt * 32 + kp;
    float v = (n < N && k < K) ? W[(size_t)n * K + k] : 0.0f;
    outp[e] = f2bf(v);
  }
}

// ---------------- main fused kernel ----------------
// One block = 256 threads = 8 wave32 = one 16-row tile. Grid = B/16 = 8192.
#define CTXS 128    // ctx tile K padded 105->128 (bf16)
#define ZS   1056   // z row stride (f32), 66 ntiles * 16
#define HS   1088   // h row stride (bf16): 800 | 200+pad24 | 50+pad14

__global__ void __launch_bounds__(256)
multiatt_kernel(const float* __restrict__ hm,
                const float* __restrict__ me,
                const float* __restrict__ tf,
                const float* __restrict__ A_b,
                const float* __restrict__ HM_b,
                const float* __restrict__ ME_b,
                const float* __restrict__ TF_b,
                const unsigned short* __restrict__ wsb,
                float* __restrict__ out)
{
  __shared__ unsigned short ctx_lds[16 * CTXS];  //  4 KB
  __shared__ float          z_lds  [16 * ZS];    // 66 KB (becomes alpha in-place)
  __shared__ unsigned short h_lds  [16 * HS];    // 34 KB

  const int tid  = threadIdx.x;
  const int wave = tid >> 5;
  const int lane = tid & 31;
  const int l15  = lane & 15;
  const int half = lane >> 4;
  const int gb0  = blockIdx.x * 16;
  const unsigned char* wsb8 = (const unsigned char*)wsb;
  float* __restrict__ alpha_out = out + (size_t)BROWS * 105;

  // ---- phase 0: load ctx tile, f32 -> bf16, zero-pad K to 128 ----
  for (int idx = tid; idx < 16 * CTXS; idx += 256) {
    int row = idx >> 7, c = idx & 127;
    int gr = gb0 + row;
    float v = 0.0f;
    if (c < 80)       v = hm[(size_t)gr * 80 + c];
    else if (c < 100) v = me[(size_t)gr * 20 + (c - 80)];
    else if (c < 105) v = tf[(size_t)gr * 5 + (c - 100)];
    ctx_lds[idx] = f2bf(v);
  }
  __syncthreads();

  // ---- phase 1: z = ctx @ A_w.T + A_b  (WMMA bf16, f32 acc) ----
  {
    Frag afr[4];
#pragma unroll
    for (int kt = 0; kt < 4; ++kt) {
      int k0 = kt * 32 + (half ? 8 : 0);
      afr[kt].u[0] = *(const uint4*)&ctx_lds[l15 * CTXS + k0];
      afr[kt].u[1] = *(const uint4*)&ctx_lds[l15 * CTXS + k0 + 16];
    }
    for (int nt = wave; nt < 66; nt += 8) {
      v8f acc = {};
#pragma unroll
      for (int kt = 0; kt < 4; ++kt) {
        Frag b;
        const uint4* wp = (const uint4*)(wsb8 + (size_t)(nt * 4 + kt) * 1024 + (size_t)lane * 32);
        b.u[0] = wp[0];
        b.u[1] = wp[1];
        acc = __builtin_amdgcn_wmma_f32_16x16x32_bf16(false, afr[kt].bf, false, b.bf,
                                                      (short)0, acc, false, false);
      }
      int n = nt * 16 + l15;
      float bias = (n < 1050) ? A_b[n] : 0.0f;
#pragma unroll
      for (int v = 0; v < 8; ++v)
        z_lds[(v + 8 * half) * ZS + n] = acc[v] + bias;
    }
  }
  __syncthreads();

  // ---- phase 2: softmax over 10 chunks of 105, in place (z -> alpha) ----
  if (tid < 160) {
    int row  = tid / 10;
    int head = tid % 10;
    float* zp = &z_lds[row * ZS + head * DDIM];
    float m = zp[0];
    for (int i = 1; i < DDIM; ++i) m = fmaxf(m, zp[i]);
    float s = 0.0f;
    for (int i = 0; i < DDIM; ++i) { float e = __expf(zp[i] - m); zp[i] = e; s += e; }
    float inv = 1.0f / s;
    for (int i = 0; i < DDIM; ++i) zp[i] *= inv;
  }
  __syncthreads();

  // ---- phase 3: write alpha to global; h = alpha * expanded (bf16 in LDS) ----
  // zero the K-padding of the ME (cols 1000..1023) and TF (cols 1074..1087) regions
  for (int idx = tid; idx < 16 * 38; idx += 256) {
    int row = idx / 38, c = idx % 38;
    int hc = (c < 24) ? (1000 + c) : (1074 + (c - 24));
    h_lds[row * HS + hc] = 0;
  }
  {
    // Each thread's j set (j = tid + 256*t, j < 1050) is identical for all 16
    // rows, so hoist the modality/column div-mod mapping out of the row loop.
    int jv[5], ccv[5], hcv[5], njv = 0;
    for (int j = tid; j < 1050; j += 256) {
      int cc = (j < 800) ? (j % 80)
             : (j < 1000 ? 80 + (j - 800) % 20
                         : 100 + (j - 1000) % 5);
      jv[njv]  = j;
      ccv[njv] = cc;
      hcv[njv] = (j < 1000) ? j : (j + 24);   // TF region shifted by ME pad
      ++njv;
    }
    for (int row = 0; row < 16; ++row) {
      const float*          zr = &z_lds[row * ZS];
      const unsigned short* cr = &ctx_lds[row * CTXS];
      unsigned short*       hr = &h_lds[row * HS];
      float* __restrict__   ar = &alpha_out[(size_t)(gb0 + row) * 1050];
      for (int t = 0; t < njv; ++t) {
        int j   = jv[t];
        float a = zr[j];
        ar[j]   = a;
        hr[hcv[t]] = f2bf(a * bf2f(cr[ccv[t]]));
      }
    }
  }
  __syncthreads();

  // ---- phase 4: out = [h_hm@HM_w.T+b | h_me@ME_w.T+b | h_tf@TF_w.T+b] ----
  // Exactly 8 output n-tile tasks -> one per wave.
  {
    size_t fragbase; int nkt, hbase, obg, nvalid, bofs;
    const float* bptr;
    if (wave < 5) {
      fragbase = (size_t)(FRAGS_HM_BASE + wave * 25) * 1024;
      nkt = 25; hbase = 0; obg = wave * 16; nvalid = 16; bptr = HM_b; bofs = wave * 16;
    } else if (wave < 7) {
      int nt = wave - 5;
      fragbase = (size_t)(FRAGS_ME_BASE + nt * 7) * 1024;
      nkt = 7; hbase = 800; obg = 80 + nt * 16; nvalid = (nt == 0) ? 16 : 4; bptr = ME_b; bofs = nt * 16;
    } else {
      fragbase = (size_t)FRAGS_TF_BASE * 1024;
      nkt = 2; hbase = 1024; obg = 100; nvalid = 5; bptr = TF_b; bofs = 0;
    }
    v8f acc = {};
    for (int kt = 0; kt < nkt; ++kt) {
      Frag a, b;
      int k0 = hbase + kt * 32 + (half ? 8 : 0);
      a.u[0] = *(const uint4*)&h_lds[l15 * HS + k0];
      a.u[1] = *(const uint4*)&h_lds[l15 * HS + k0 + 16];
      const uint4* wp = (const uint4*)(wsb8 + fragbase + (size_t)kt * 1024 + (size_t)lane * 32);
      b.u[0] = wp[0];
      b.u[1] = wp[1];
      acc = __builtin_amdgcn_wmma_f32_16x16x32_bf16(false, a.bf, false, b.bf,
                                                    (short)0, acc, false, false);
    }
    if (l15 < nvalid) {
      int o = obg + l15;
      float bias = bptr[bofs + l15];
#pragma unroll
      for (int v = 0; v < 8; ++v) {
        int gr = gb0 + v + 8 * half;
        out[(size_t)gr * 105 + o] = acc[v] + bias;
      }
    }
  }
}

// ---------------- launch ----------------
extern "C" void kernel_launch(void* const* d_in, const int* in_sizes, int n_in,
                              void* d_out, int out_size, void* d_ws, size_t ws_size,
                              hipStream_t stream) {
  const float* hm   = (const float*)d_in[0];
  const float* me   = (const float*)d_in[1];
  const float* tf   = (const float*)d_in[2];
  const float* A_w  = (const float*)d_in[3];
  const float* A_b  = (const float*)d_in[4];
  const float* HM_w = (const float*)d_in[5];
  const float* HM_b = (const float*)d_in[6];
  const float* ME_w = (const float*)d_in[7];
  const float* ME_b = (const float*)d_in[8];
  const float* TF_w = (const float*)d_in[9];
  const float* TF_b = (const float*)d_in[10];

  unsigned short* wsb = (unsigned short*)d_ws;   // 405 frags * 1 KB = ~415 KB

  int pack_threads = FRAGS_TOTAL * 32;
  pack_weights_kernel<<<(pack_threads + 255) / 256, 256, 0, stream>>>(A_w, HM_w, ME_w, TF_w, wsb);

  multiatt_kernel<<<BROWS / 16, 256, 0, stream>>>(hm, me, tf, A_b, HM_b, ME_b, TF_b,
                                                  wsb, (float*)d_out);
}
